// DualEncoderGraphModel_5188320493830
// MI455X (gfx1250) — matrix operand order repaired and checked by reading, not compiled, not run on hardware
//
#include <hip/hip_runtime.h>
#include <hip/hip_bf16.h>
#include <math.h>

#define N_NODES  32768
#define N_EDGES  262144
#define N_GRAPHS 512
#define IN_DIM   300
#define HID      128
#define HEADS    4
#define D_FUSED  512
#define L_LAYERS 3

typedef __bf16 bf16x16 __attribute__((ext_vector_type(16)));
typedef __bf16 bf16x8  __attribute__((ext_vector_type(8)));
typedef float  f32x8   __attribute__((ext_vector_type(8)));
typedef float  f32x4   __attribute__((ext_vector_type(4)));

__device__ __forceinline__ unsigned short f2bf(float f) {
  unsigned u = __builtin_bit_cast(unsigned, f);
  u += 0x7FFFu + ((u >> 16) & 1u);           // round-to-nearest-even
  return (unsigned short)(u >> 16);
}
// monotone encoding of float into uint for atomicMax-based segment max
__device__ __forceinline__ unsigned enc_f32(float f) {
  unsigned u = __builtin_bit_cast(unsigned, f);
  return (u & 0x80000000u) ? ~u : (u | 0x80000000u);
}
__device__ __forceinline__ float dec_f32(unsigned e) {
  unsigned u = (e & 0x80000000u) ? (e & 0x7FFFFFFFu) : ~e;
  return __builtin_bit_cast(float, u);
}

__device__ __forceinline__ bf16x16 load_frag16(const __bf16* lo_p, const __bf16* hi_p) {
  bf16x8 lo = *(const bf16x8*)lo_p;
  bf16x8 hi = *(const bf16x8*)hi_p;
  return __builtin_shufflevector(lo, hi, 0,1,2,3,4,5,6,7,8,9,10,11,12,13,14,15);
}

// ---------------- encoder: t[n][0:128]=relu(x@syn_w+b), t[n][128:256]=relu(x@ant_w+b)
__global__ void __launch_bounds__(256) encode_nodes_kernel(
    const float* __restrict__ x,
    const float* __restrict__ syn_w, const float* __restrict__ syn_b,
    const float* __restrict__ ant_w, const float* __restrict__ ant_b,
    float* __restrict__ t)
{
  __shared__ float xs[IN_DIM];
  int n = blockIdx.x;
  int j = threadIdx.x;  // 0..255
  for (int i = threadIdx.x; i < IN_DIM; i += blockDim.x)
    xs[i] = x[(size_t)n * IN_DIM + i];
  __syncthreads();
  const float* w;
  float b;
  int col;
  if (j < HID) { w = syn_w; b = syn_b[j]; col = j; }
  else         { w = ant_w; b = ant_b[j - HID]; col = j - HID; }
  float acc = 0.f;
  for (int i = 0; i < IN_DIM; ++i)
    acc = fmaf(xs[i], w[(size_t)i * HID + col], acc);
  acc += b;
  t[(size_t)n * 256 + j] = fmaxf(acc, 0.f);
}

// ---------------- f32 -> bf16 convert
__global__ void f32_to_bf16_kernel(const float* __restrict__ in,
                                   unsigned short* __restrict__ out, int n) {
  int i = blockIdx.x * blockDim.x + threadIdx.x;
  if (i < n) out[i] = f2bf(in[i]);
}

// ---------------- W[K,N] (f32) -> Wt[N,K] (bf16)
__global__ void transpose_bf16_kernel(const float* __restrict__ W,
                                      unsigned short* __restrict__ Wt, int K, int Nc) {
  int idx = blockIdx.x * blockDim.x + threadIdx.x;
  if (idx >= K * Nc) return;
  int k = idx % K, n = idx / K;
  Wt[(size_t)n * K + k] = f2bf(W[(size_t)k * Nc + n]);
}

// ================= WMMA GEMM =================
// C[M,N] = A[M,K](bf16) * Bt[N,K]^T(bf16) + bias, optional relu.
// Each wave computes a 64(M) x 32(N) output block (4x2 accumulator tiles).
// Per K=32 step: 4 A-fragments + 2 B-fragments feed 8 v_wmma_f32_16x16x32_bf16.
// Single-stage software pipeline: fragments for step k+32 are loaded before the
// WMMAs of step k execute, so WMMAs never wait on just-issued loads.

__device__ __forceinline__ void load_set(const __bf16* pa, const __bf16* pb0,
                                         const __bf16* pb1, int K, int k,
                                         bf16x16 a[4], bf16x16 b[2]) {
#pragma unroll
  for (int r = 0; r < 4; ++r) {
    const __bf16* p = pa + (size_t)r * 16 * K + k;
    a[r] = load_frag16(p, p + 16);           // K halves kk..+7 / kk+16..+23 (per-lane sel)
  }
  b[0] = load_frag16(pb0 + k, pb0 + k + 16); // K halves kk..+7 / kk+16..+23 (per-lane sel)
  b[1] = load_frag16(pb1 + k, pb1 + k + 16);
}

__device__ __forceinline__ void wmma_set(const bf16x16 a[4], const bf16x16 b[2],
                                         f32x8 acc[4][2]) {
#pragma unroll
  for (int r = 0; r < 4; ++r)
#pragma unroll
    for (int c = 0; c < 2; ++c)
      acc[r][c] = __builtin_amdgcn_wmma_f32_16x16x32_bf16(
          false, a[r], false, b[c], (short)0, acc[r][c], false, false);
}

__global__ void __launch_bounds__(256) gemm_bf16_wmma_kernel(
    const unsigned short* __restrict__ A,    // [M,K] bf16 row-major
    const unsigned short* __restrict__ Bt,   // [N,K] bf16 (B transposed)
    const float* __restrict__ bias,          // [N]
    float* __restrict__ C,                   // [M,N] f32
    int M, int Nn, int K, int relu)          // K must be a multiple of 64
{
  int wave = blockIdx.x * (blockDim.x >> 5) + (threadIdx.x >> 5);
  int lane = threadIdx.x & 31;
  int tiles_n = Nn >> 5;            // 32-wide column blocks
  int total = (M >> 6) * tiles_n;   // 64-tall row blocks
  if (wave >= total) return;        // whole-wave uniform -> EXEC all ones at WMMA
  int tm = wave / tiles_n;
  int tn = wave % tiles_n;

  int hrow = lane & 15;   // A row / B col within 16-tile
  int hsel = lane >> 4;   // K-half select

  const __bf16* pa  = (const __bf16*)A  + (size_t)(tm * 64 + hrow) * K + hsel * 8;
  const __bf16* pb0 = (const __bf16*)Bt + (size_t)(tn * 32 + hrow) * K + hsel * 8;
  const __bf16* pb1 = pb0 + (size_t)16 * K;

  f32x8 acc[4][2];
#pragma unroll
  for (int r = 0; r < 4; ++r)
#pragma unroll
    for (int c = 0; c < 2; ++c)
      acc[r][c] = (f32x8){0.f, 0.f, 0.f, 0.f, 0.f, 0.f, 0.f, 0.f};

  bf16x16 aP[4], bP[2];   // ping
  bf16x16 aQ[4], bQ[2];   // pong
  load_set(pa, pb0, pb1, K, 0, aP, bP);
  int k = 0;
  for (; k < K - 64; k += 64) {
    load_set(pa, pb0, pb1, K, k + 32, aQ, bQ);  // prefetch next set
    wmma_set(aP, bP, acc);                      // consume set loaded last step
    load_set(pa, pb0, pb1, K, k + 64, aP, bP);
    wmma_set(aQ, bQ, acc);
  }
  load_set(pa, pb0, pb1, K, K - 32, aQ, bQ);    // k == K-64 here
  wmma_set(aP, bP, acc);
  wmma_set(aQ, bQ, acc);

#pragma unroll
  for (int c = 0; c < 2; ++c) {
    int col = tn * 32 + c * 16 + (lane & 15);
    float bb = bias ? bias[col] : 0.f;
#pragma unroll
    for (int r4 = 0; r4 < 4; ++r4) {
#pragma unroll
      for (int r = 0; r < 8; ++r) {
        int m = tm * 64 + r4 * 16 + r + hsel * 8;  // C layout: lanes 0-15 M=r, 16-31 M=r+8
        float v = acc[r4][c][r] + bb;
        if (relu) v = fmaxf(v, 0.f);
        C[(size_t)m * Nn + col] = v;
      }
    }
  }
}

// ---------------- per-edge attention logits + segment max (wave per edge)
__global__ void __launch_bounds__(256) edge_logits_kernel(
    const float* __restrict__ Q, const float* __restrict__ Kf,
    const int* __restrict__ src, const int* __restrict__ dst,
    float* __restrict__ logits, unsigned* __restrict__ menc, int E)
{
  int lane = threadIdx.x & 31;
  int e = blockIdx.x * (blockDim.x >> 5) + (threadIdx.x >> 5);
  if (e >= E) return;
  int s = src[e], d = dst[e];
  const float* q = Q + (size_t)d * D_FUSED + lane * 16;
  const float* k = Kf + (size_t)s * D_FUSED + lane * 16;
  float p = 0.f;
#pragma unroll
  for (int i = 0; i < 16; i += 4) {
    f32x4 qa = *(const f32x4*)(q + i);
    f32x4 ka = *(const f32x4*)(k + i);
    p = fmaf(qa[0], ka[0], p);
    p = fmaf(qa[1], ka[1], p);
    p = fmaf(qa[2], ka[2], p);
    p = fmaf(qa[3], ka[3], p);
  }
  // reduce within 8-lane groups (one head per group: 128 dims = 8 lanes * 16)
  p += __shfl_xor(p, 1);
  p += __shfl_xor(p, 2);
  p += __shfl_xor(p, 4);
  if ((lane & 7) == 0) {
    int h = lane >> 3;
    float lg = p * 0.08838834764831845f;  // 1/sqrt(128)
    logits[(size_t)e * HEADS + h] = lg;
    atomicMax(&menc[(size_t)d * HEADS + h], enc_f32(lg));
  }
}

// ---------------- e = exp(logit - m[dst]); z[dst] += e
__global__ void edge_exp_kernel(float* __restrict__ logits,
                                const unsigned* __restrict__ menc,
                                const int* __restrict__ dst,
                                float* __restrict__ z, int E)
{
  int idx = blockIdx.x * blockDim.x + threadIdx.x;
  if (idx >= E * HEADS) return;
  int e = idx >> 2, h = idx & 3;
  int d = dst[e];
  float m = dec_f32(menc[(size_t)d * HEADS + h]);
  float ev = __expf(logits[idx] - m);
  logits[idx] = ev;
  atomicAdd(&z[(size_t)d * HEADS + h], ev);
}

// ---------------- out[dst] += alpha * V[src]   (wave per edge)
__global__ void __launch_bounds__(256) edge_scatter_kernel(
    const float* __restrict__ ebuf, const float* __restrict__ z,
    const float* __restrict__ V, const int* __restrict__ src,
    const int* __restrict__ dst, float* __restrict__ out, int E)
{
  int lane = threadIdx.x & 31;
  int e = blockIdx.x * (blockDim.x >> 5) + (threadIdx.x >> 5);
  if (e >= E) return;
  int s = src[e], d = dst[e];
  int h = lane >> 3;
  float alpha = ebuf[(size_t)e * HEADS + h] / (z[(size_t)d * HEADS + h] + 1e-16f);
  const float* vp = V + (size_t)s * D_FUSED + lane * 16;
  float* op = out + (size_t)d * D_FUSED + lane * 16;
#pragma unroll
  for (int i = 0; i < 16; ++i)
    atomicAdd(op + i, alpha * vp[i]);
}

// ---------------- h = relu(h + S)
__global__ void add_relu_kernel(float* __restrict__ h, const float* __restrict__ S, int n) {
  int i = blockIdx.x * blockDim.x + threadIdx.x;
  if (i < n) h[i] = fmaxf(h[i] + S[i], 0.f);
}

// ---------------- pooling
__global__ void pool_sum_kernel(const float* __restrict__ h, const int* __restrict__ batch,
                                float* __restrict__ pooled, int Nn) {
  int idx = blockIdx.x * blockDim.x + threadIdx.x;
  if (idx >= Nn * D_FUSED) return;
  int n = idx >> 9, j = idx & 511;
  atomicAdd(&pooled[(size_t)batch[n] * D_FUSED + j], h[idx]);
}
__global__ void pool_cnt_kernel(const int* __restrict__ batch, float* __restrict__ cnt, int Nn) {
  int i = blockIdx.x * blockDim.x + threadIdx.x;
  if (i < Nn) atomicAdd(&cnt[batch[i]], 1.0f);
}
__global__ void pool_div_kernel(float* __restrict__ pooled, const float* __restrict__ cnt, int total) {
  int idx = blockIdx.x * blockDim.x + threadIdx.x;
  if (idx < total) pooled[idx] /= fmaxf(cnt[idx >> 9], 1.0f);
}

// ---------------- final head: sigmoid(hid @ w2 + b2)
__global__ void classifier_out_kernel(const float* __restrict__ hid,
                                      const float* __restrict__ w2,
                                      const float* __restrict__ b2,
                                      float* __restrict__ out, int Gn) {
  int g = blockIdx.x * blockDim.x + threadIdx.x;
  if (g >= Gn) return;
  float acc = b2[0];
  for (int j = 0; j < HID; ++j)
    acc = fmaf(hid[(size_t)g * HID + j], w2[j], acc);
  out[g] = 1.f / (1.f + __expf(-acc));
}

extern "C" void kernel_launch(void* const* d_in, const int* in_sizes, int n_in,
                              void* d_out, int out_size, void* d_ws, size_t ws_size,
                              hipStream_t stream) {
  const float* x        = (const float*)d_in[0];
  const int*   eidx     = (const int*)d_in[1];
  const int*   batch    = (const int*)d_in[2];
  const float* syn_w    = (const float*)d_in[3];
  const float* syn_b    = (const float*)d_in[4];
  const float* ant_w    = (const float*)d_in[5];
  const float* ant_b    = (const float*)d_in[6];
  const float* fusion_w = (const float*)d_in[7];
  const float* fusion_b = (const float*)d_in[8];
  const float* Wq       = (const float*)d_in[9];
  const float* bq       = (const float*)d_in[10];
  const float* Wk       = (const float*)d_in[11];
  const float* bk       = (const float*)d_in[12];
  const float* Wv       = (const float*)d_in[13];
  const float* bv       = (const float*)d_in[14];
  const float* Wskip    = (const float*)d_in[15];
  const float* bskip    = (const float*)d_in[16];
  const float* cls_w1   = (const float*)d_in[17];
  const float* cls_b1   = (const float*)d_in[18];
  const float* cls_w2   = (const float*)d_in[19];
  const float* cls_b2   = (const float*)d_in[20];
  float* out = (float*)d_out;

  const int* src = eidx;
  const int* dst = eidx + N_EDGES;

  // ---- workspace carve-up (all 256B aligned)
  char* ws = (char*)d_ws;
  size_t off = 0;
  auto carve = [&](size_t bytes) -> char* {
    char* p = ws + off;
    off = (off + bytes + 255) & ~(size_t)255;
    return p;
  };
  float*          h      = (float*)carve((size_t)N_NODES * D_FUSED * 4);   // node features / attn accum
  unsigned short* hbf    = (unsigned short*)carve((size_t)N_NODES * D_FUSED * 2);
  float*          Qb     = (float*)carve((size_t)N_NODES * D_FUSED * 4);   // also encoder t (N x 256)
  float*          Kb     = (float*)carve((size_t)N_NODES * D_FUSED * 4);
  float*          Vb     = (float*)carve((size_t)N_NODES * D_FUSED * 4);
  float*          Sb     = (float*)carve((size_t)N_NODES * D_FUSED * 4);
  unsigned short* Wt     = (unsigned short*)carve((size_t)D_FUSED * D_FUSED * 2);
  float*          elog   = (float*)carve((size_t)N_EDGES * HEADS * 4);     // logits -> exp values
  unsigned*       menc   = (unsigned*)carve((size_t)N_NODES * HEADS * 4);
  float*          zbuf   = (float*)carve((size_t)N_NODES * HEADS * 4);
  float*          pooled = (float*)carve((size_t)N_GRAPHS * D_FUSED * 4);
  float*          cnt    = (float*)carve((size_t)N_GRAPHS * 4);
  float*          hid    = (float*)carve((size_t)N_GRAPHS * HID * 4);
  unsigned short* pbf    = (unsigned short*)carve((size_t)N_GRAPHS * D_FUSED * 2);
  (void)ws_size; (void)in_sizes; (void)n_in; (void)out_size;

  const int TB = 256;
  auto grid1 = [](int n, int tb) { return (n + tb - 1) / tb; };
  auto gemm_grid = [](int M, int Nn) { return ((M >> 6) * (Nn >> 5) + 7) / 8; };

  // ===== Encoder: t = [relu(x@syn_w+b) | relu(x@ant_w+b)]  -> fusion GEMM -> h
  float* t = Qb;  // N x 256, fits in Q region
  encode_nodes_kernel<<<N_NODES, TB, 0, stream>>>(x, syn_w, syn_b, ant_w, ant_b, t);
  f32_to_bf16_kernel<<<grid1(N_NODES * 256, TB), TB, 0, stream>>>(t, hbf, N_NODES * 256);
  transpose_bf16_kernel<<<grid1(256 * D_FUSED, TB), TB, 0, stream>>>(fusion_w, Wt, 256, D_FUSED);
  gemm_bf16_wmma_kernel<<<gemm_grid(N_NODES, D_FUSED), TB, 0, stream>>>(
      hbf, Wt, fusion_b, h, N_NODES, D_FUSED, 256, 0);

  // ===== 3 TransformerConv layers
  for (int l = 0; l < L_LAYERS; ++l) {
    size_t wo = (size_t)l * D_FUSED * D_FUSED;
    size_t bo = (size_t)l * D_FUSED;
    f32_to_bf16_kernel<<<grid1(N_NODES * D_FUSED, TB), TB, 0, stream>>>(h, hbf, N_NODES * D_FUSED);

    transpose_bf16_kernel<<<grid1(D_FUSED * D_FUSED, TB), TB, 0, stream>>>(Wq + wo, Wt, D_FUSED, D_FUSED);
    gemm_bf16_wmma_kernel<<<gemm_grid(N_NODES, D_FUSED), TB, 0, stream>>>(
        hbf, Wt, bq + bo, Qb, N_NODES, D_FUSED, D_FUSED, 0);
    transpose_bf16_kernel<<<grid1(D_FUSED * D_FUSED, TB), TB, 0, stream>>>(Wk + wo, Wt, D_FUSED, D_FUSED);
    gemm_bf16_wmma_kernel<<<gemm_grid(N_NODES, D_FUSED), TB, 0, stream>>>(
        hbf, Wt, bk + bo, Kb, N_NODES, D_FUSED, D_FUSED, 0);
    transpose_bf16_kernel<<<grid1(D_FUSED * D_FUSED, TB), TB, 0, stream>>>(Wv + wo, Wt, D_FUSED, D_FUSED);
    gemm_bf16_wmma_kernel<<<gemm_grid(N_NODES, D_FUSED), TB, 0, stream>>>(
        hbf, Wt, bv + bo, Vb, N_NODES, D_FUSED, D_FUSED, 0);
    transpose_bf16_kernel<<<grid1(D_FUSED * D_FUSED, TB), TB, 0, stream>>>(Wskip + wo, Wt, D_FUSED, D_FUSED);
    gemm_bf16_wmma_kernel<<<gemm_grid(N_NODES, D_FUSED), TB, 0, stream>>>(
        hbf, Wt, bskip + bo, Sb, N_NODES, D_FUSED, D_FUSED, 0);

    hipMemsetAsync(menc, 0, (size_t)N_NODES * HEADS * 4, stream);  // encoded 0 < every real logit
    hipMemsetAsync(zbuf, 0, (size_t)N_NODES * HEADS * 4, stream);
    edge_logits_kernel<<<N_EDGES / 8, TB, 0, stream>>>(Qb, Kb, src, dst, elog, menc, N_EDGES);
    edge_exp_kernel<<<grid1(N_EDGES * HEADS, TB), TB, 0, stream>>>(elog, menc, dst, zbuf, N_EDGES);
    hipMemsetAsync(h, 0, (size_t)N_NODES * D_FUSED * 4, stream);   // reuse h as message accumulator
    edge_scatter_kernel<<<N_EDGES / 8, TB, 0, stream>>>(elog, zbuf, Vb, src, dst, h, N_EDGES);
    add_relu_kernel<<<grid1(N_NODES * D_FUSED, TB), TB, 0, stream>>>(h, Sb, N_NODES * D_FUSED);
  }

  // ===== mean pool per graph
  hipMemsetAsync(pooled, 0, (size_t)N_GRAPHS * D_FUSED * 4, stream);
  hipMemsetAsync(cnt, 0, (size_t)N_GRAPHS * 4, stream);
  pool_sum_kernel<<<grid1(N_NODES * D_FUSED, TB), TB, 0, stream>>>(h, batch, pooled, N_NODES);
  pool_cnt_kernel<<<grid1(N_NODES, TB), TB, 0, stream>>>(batch, cnt, N_NODES);
  pool_div_kernel<<<grid1(N_GRAPHS * D_FUSED, TB), TB, 0, stream>>>(pooled, cnt, N_GRAPHS * D_FUSED);

  // ===== classifier: relu(pooled @ cls_w1 + b1) via WMMA, then sigmoid head
  f32_to_bf16_kernel<<<grid1(N_GRAPHS * D_FUSED, TB), TB, 0, stream>>>(pooled, pbf, N_GRAPHS * D_FUSED);
  transpose_bf16_kernel<<<grid1(D_FUSED * HID, TB), TB, 0, stream>>>(cls_w1, Wt, D_FUSED, HID);
  gemm_bf16_wmma_kernel<<<gemm_grid(N_GRAPHS, HID), TB, 0, stream>>>(
      pbf, Wt, cls_b1, hid, N_GRAPHS, HID, D_FUSED, 1);
  classifier_out_kernel<<<grid1(N_GRAPHS, TB), TB, 0, stream>>>(hid, cls_w2, cls_b2, out, N_GRAPHS);
}
